// EdgeNetwork_59528246723026
// MI455X (gfx1250) — matrix operand order repaired and checked by reading, not compiled, not run on hardware
//
#include <hip/hip_runtime.h>
#include <hip/hip_bf16.h>

typedef __attribute__((ext_vector_type(16))) _Float16 v16h;
typedef __attribute__((ext_vector_type(8)))  float    v8f;
typedef __attribute__((ext_vector_type(4)))  _Float16 h4;
typedef __attribute__((ext_vector_type(2)))  _Float16 h2;

#define BLOCK   256
#define NWAVES  8
#define EPB     (NWAVES * 16)   // 128 edges per block-tile
#define NFRAG   32              // 16 (L1) + 8 (L2) + 8 (L3) pre-swizzled A fragments
#define PBLOCKS 2048            // persistent blocks (grid-stride over tiles)

// Native gfx1250 v_tanh_f32 when the toolchain exposes it; else v_exp_f32 path.
__device__ __forceinline__ float fast_tanh(float x) {
#if __has_builtin(__builtin_amdgcn_tanhf)
  return __builtin_amdgcn_tanhf(x);
#elif __has_builtin(__builtin_amdgcn_tanh_f32)
  return __builtin_amdgcn_tanh_f32(x);
#else
  float t = __expf(2.0f * x);
  return __fdividef(t - 1.0f, t + 1.0f);
#endif
}

// Load 16 contiguous f32 and convert to a 16-f16 B-fragment lane vector.
__device__ __forceinline__ v16h load_run(const float* __restrict__ p) {
  union { v16h v; h4 h[4]; } r;
#pragma unroll
  for (int j = 0; j < 4; ++j) {
    float4 q = *(const float4*)(p + 4 * j);
    h4 t;
    t[0] = (_Float16)q.x; t[1] = (_Float16)q.y;
    t[2] = (_Float16)q.z; t[3] = (_Float16)q.w;
    r.h[j] = t;
  }
  return r.v;
}

// acc[t][r] = bias[16t + 8hf + r]  (D' layout: M = out-feature, N = edge)
__device__ __forceinline__ void init_bias(v8f* acc, const float* __restrict__ bArr, int hf) {
#pragma unroll
  for (int t = 0; t < 4; ++t)
#pragma unroll
    for (int r = 0; r < 8; ++r) acc[t][r] = bArr[16 * t + 8 * hf + r];
}

// LayerNorm + affine + tanh over the 64 features of each edge (column n).
// Lane (n,hf) holds features {16t + 8hf + r}; partner half-wave holds the rest.
__device__ __forceinline__ void ln_tanh_t(v8f* acc, const float* __restrict__ gA,
                                          const float* __restrict__ bA, int hf) {
  float s = 0.0f, q = 0.0f;
#pragma unroll
  for (int t = 0; t < 4; ++t)
#pragma unroll
    for (int r = 0; r < 8; ++r) { float v = acc[t][r]; s += v; q += v * v; }
  s += __shfl_xor(s, 16, 32);
  q += __shfl_xor(q, 16, 32);
  float mean = s * (1.0f / 64.0f);
  float rsig = rsqrtf(q * (1.0f / 64.0f) - mean * mean + 1e-5f);
#pragma unroll
  for (int t = 0; t < 4; ++t) {
#pragma unroll
    for (int r = 0; r < 8; ++r) {
      int   fi = 16 * t + 8 * hf + r;
      float rg = rsig * gA[fi];
      acc[t][r] = fast_tanh(fmaf(acc[t][r], rg, fmaf(-mean, rg, bA[fi])));
    }
  }
}

// Convert D' accumulators (f32) into next layer's B fragments (f16) in-register:
// pack row-pairs with cvt_pk, swap dword groups with the partner half-wave.
__device__ __forceinline__ void pack_to_B(const v8f* acc, v16h* B, int hf) {
  unsigned pk[4][4];
#pragma unroll
  for (int t = 0; t < 4; ++t)
#pragma unroll
    for (int d = 0; d < 4; ++d) {
      union { unsigned u; h2 h; } cv;
      cv.h[0] = (_Float16)acc[t][2 * d];
      cv.h[1] = (_Float16)acc[t][2 * d + 1];
      pk[t][d] = cv.u;
    }
#pragma unroll
  for (int kc = 0; kc < 2; ++kc) {
    union { v16h v; unsigned u[8]; } bb;
#pragma unroll
    for (int d = 0; d < 4; ++d) {
      unsigned selfD = hf ? pk[2 * kc + 1][d] : pk[2 * kc][d];     // tile I need
      unsigned sendD = hf ? pk[2 * kc][d]     : pk[2 * kc + 1][d]; // tile partner needs
      unsigned rv = (unsigned)__shfl_xor((int)sendD, 16, 32);
      bb.u[d]     = hf ? rv : selfD;   // B elements 0..7  (K = 16*hf + 0..7 of chunk)
      bb.u[4 + d] = hf ? selfD : rv;   // B elements 8..15 (K = 16*hf + 8..15 of chunk)
    }
    B[kc] = bb.v;
  }
}

__global__ __launch_bounds__(BLOCK, 4)
void edge_mlp_kernel(const float* __restrict__ x,
                     const long long* __restrict__ eidx,   // [2, nEdges] int64
                     const float* __restrict__ W1, const float* __restrict__ b1,
                     const float* __restrict__ g1, const float* __restrict__ bt1,
                     const float* __restrict__ W2, const float* __restrict__ b2,
                     const float* __restrict__ g2, const float* __restrict__ bt2,
                     const float* __restrict__ W3, const float* __restrict__ b3,
                     const float* __restrict__ g3, const float* __restrict__ bt3,
                     const float* __restrict__ W4, const float* __restrict__ b4,
                     float* __restrict__ out, int nEdges, int nTiles) {
  // Pre-swizzled W^T A-fragments: frag f, lane ln -> 16 contiguous f16 (32B aligned)
  __shared__ __align__(64) _Float16 Wswz[NFRAG * 512];
  __shared__ __align__(16) float    cvec[10 * 64 + 4]; // b1,g1,bt1,b2,g2,bt2,b3,g3,bt3,w4,b4

  const int tid = threadIdx.x;

  // ---- stage weights into A-fragment layout (16-bit A 16x32, contiguous k = 16*half + i), once ----
  for (int idx = tid; idx < NFRAG * 512; idx += BLOCK) {
    int f   = idx >> 9;
    int rem = idx & 511;
    int ln  = rem >> 4;
    int i   = rem & 15;
    int mm  = ln & 15;
    int hfa = ln >> 4;
    int k   = i + 16 * hfa;
    const float* Wp; int g;
    if (f < 16)      { Wp = W1; g = f; }
    else if (f < 24) { Wp = W2; g = f - 16; }
    else             { Wp = W3; g = f - 24; }
    int kc = g >> 2, t = g & 3;
    // A[m][k] = W[32kc + k][16t + m]  (W row-major [fan_in][64])
    Wswz[idx] = (_Float16)Wp[(32 * kc + k) * 64 + 16 * t + mm];
  }
  if (tid < 64) {
    cvec[tid]           = b1[tid];
    cvec[1 * 64 + tid]  = g1[tid];
    cvec[2 * 64 + tid]  = bt1[tid];
    cvec[3 * 64 + tid]  = b2[tid];
    cvec[4 * 64 + tid]  = g2[tid];
    cvec[5 * 64 + tid]  = bt2[tid];
    cvec[6 * 64 + tid]  = b3[tid];
    cvec[7 * 64 + tid]  = g3[tid];
    cvec[8 * 64 + tid]  = bt3[tid];
    cvec[9 * 64 + tid]  = W4[tid];
  }
  if (tid == 0) cvec[640] = b4[0];
  __syncthreads();

  const v16h* __restrict__ Afrag = (const v16h*)Wswz;

  const int lane = tid & 31;
  const int n    = lane & 15;   // edge column
  const int hf   = lane >> 4;   // half-wave select
  const int wave = tid >> 5;

  // ---- persistent grid-stride loop over 128-edge tiles ----
  for (int tile = blockIdx.x; tile < nTiles; tile += gridDim.x) {
    // Keep LDS fragment loads inside the loop (2x ds_load_b128 each) instead of
    // letting LICM hoist 32 fragments into VGPRs and spill to scratch.
    asm volatile("" ::: "memory");

    const long long eBase = (long long)tile * EPB + (long long)wave * 16;
    long long ej = eBase + n;
    if (ej >= nEdges) ej = nEdges - 1;

    // gather: B fragments straight from global (K = features in-lane)
    const long long sN = eidx[ej];
    const long long eN = eidx[nEdges + ej];
    const float* __restrict__ rs = x + (size_t)sN * 64;
    const float* __restrict__ re = x + (size_t)eN * 64;

    v16h B[4];
    B[0] = load_run(rs + 16 * hf);        // K =   0..31  (chunk 0)
    B[1] = load_run(rs + 32 + 16 * hf);   // K =  32..63
    B[2] = load_run(re + 16 * hf);        // K =  64..95
    B[3] = load_run(re + 32 + 16 * hf);   // K =  96..127

    v8f acc[4];

    // layer 1: D' = W1^T (64x128) x e^T (128x16), 4 K-chunks x 4 M-tiles
    init_bias(acc, &cvec[0], hf);
#pragma unroll
    for (int kc = 0; kc < 4; ++kc) {
#pragma unroll
      for (int t = 0; t < 4; ++t) {
        v16h a = Afrag[(kc * 4 + t) * 32 + lane];
        acc[t] = __builtin_amdgcn_wmma_f32_16x16x32_f16(false, a, false, B[kc],
                                                        (short)0, acc[t], false, false);
      }
    }
    ln_tanh_t(acc, &cvec[1 * 64], &cvec[2 * 64], hf);
    pack_to_B(acc, B, hf);

    // layer 2
    init_bias(acc, &cvec[3 * 64], hf);
#pragma unroll
    for (int kc = 0; kc < 2; ++kc) {
#pragma unroll
      for (int t = 0; t < 4; ++t) {
        v16h a = Afrag[(16 + kc * 4 + t) * 32 + lane];
        acc[t] = __builtin_amdgcn_wmma_f32_16x16x32_f16(false, a, false, B[kc],
                                                        (short)0, acc[t], false, false);
      }
    }
    ln_tanh_t(acc, &cvec[4 * 64], &cvec[5 * 64], hf);
    pack_to_B(acc, B, hf);

    // layer 3
    init_bias(acc, &cvec[6 * 64], hf);
#pragma unroll
    for (int kc = 0; kc < 2; ++kc) {
#pragma unroll
      for (int t = 0; t < 4; ++t) {
        v16h a = Afrag[(24 + kc * 4 + t) * 32 + lane];
        acc[t] = __builtin_amdgcn_wmma_f32_16x16x32_f16(false, a, false, B[kc],
                                                        (short)0, acc[t], false, false);
      }
    }
    ln_tanh_t(acc, &cvec[7 * 64], &cvec[8 * 64], hf);

    // layer 4: per-edge dot with W4 + sigmoid
    float ps = 0.0f;
#pragma unroll
    for (int t = 0; t < 4; ++t)
#pragma unroll
      for (int r = 0; r < 8; ++r)
        ps += acc[t][r] * cvec[9 * 64 + 16 * t + 8 * hf + r];
    ps += __shfl_xor(ps, 16, 32);
    float o = __fdividef(1.0f, 1.0f + __expf(-(ps + cvec[640])));
    if (hf == 0 && eBase + n < (long long)nEdges) out[eBase + n] = o;
  }
}

extern "C" void kernel_launch(void* const* d_in, const int* in_sizes, int n_in,
                              void* d_out, int out_size, void* d_ws, size_t ws_size,
                              hipStream_t stream) {
  const float*     x    = (const float*)d_in[0];
  const long long* eidx = (const long long*)d_in[1];
  const float* W1  = (const float*)d_in[2];
  const float* b1  = (const float*)d_in[3];
  const float* g1  = (const float*)d_in[4];
  const float* bt1 = (const float*)d_in[5];
  const float* W2  = (const float*)d_in[6];
  const float* b2  = (const float*)d_in[7];
  const float* g2  = (const float*)d_in[8];
  const float* bt2 = (const float*)d_in[9];
  const float* W3  = (const float*)d_in[10];
  const float* b3  = (const float*)d_in[11];
  const float* g3  = (const float*)d_in[12];
  const float* bt3 = (const float*)d_in[13];
  const float* W4  = (const float*)d_in[14];
  const float* b4  = (const float*)d_in[15];

  const int nEdges = in_sizes[1] / 2;
  const int nTiles = (nEdges + EPB - 1) / EPB;
  const int blocks = nTiles < PBLOCKS ? nTiles : PBLOCKS;

  edge_mlp_kernel<<<blocks, BLOCK, 0, stream>>>(
      x, eidx, W1, b1, g1, bt1, W2, b2, g2, bt2, W3, b3, g3, bt3, W4, b4,
      (float*)d_out, nEdges, nTiles);
}
